// Model_39676907884055
// MI455X (gfx1250) — matrix operand order repaired and checked by reading, not compiled, not run on hardware
//
#include <hip/hip_runtime.h>

// ---------------------------------------------------------------------------
// Flash-attention for B=16, S=2048, D=768 fp32 on MI455X (gfx1250, wave32).
// Compute-bound (~515 flop/byte) -> both GEMMs on v_wmma_f32_16x16x32_bf16.
// Block = 256 threads (8 waves), BM=32 queries, BN=64 keys per iteration.
//   QK : 8 waves x (16x16 tile, 24 WMMA over D=768)
//   PV : 8 waves = 2 (M) x 4 (192-dim slice), 12 acc tiles, 24 WMMA/iter
// fp32->bf16 staging uses single v_perm_b32 per element pair (truncation).
// LDS: Q(bf16) + K(bf16 row-major) + V^T(bf16) + probs + scores (~274 KB).
// ---------------------------------------------------------------------------

typedef __attribute__((ext_vector_type(16))) __bf16 v16bf;
typedef __attribute__((ext_vector_type(8)))  __bf16 v8bf;
typedef __attribute__((ext_vector_type(8)))  float  v8f;

#define B_   16
#define S_   2048
#define D_   768
#define BM   32
#define BN   64
#define QLD  (D_ + 8)   // 776, keeps 16B alignment (mult of 8 elems)
#define KLD  (D_ + 8)
#define VTLD (BN + 8)   // 72
#define PLD  (BN + 8)
#define SLD  (BN + 8)
#define NKT  (S_ / BN)  // 32 key-tile iterations
#define NDK  (D_ / 32)  // 24 WMMA k-steps for QK
#define NTD  12         // 192/16 output tiles per wave in PV
#define THREADS 256

#define CAT16(lo, hi) __builtin_shufflevector(lo, hi, 0,1,2,3,4,5,6,7,8,9,10,11,12,13,14,15)

// Pack two fp32 into a bf16 pair (truncation) with one v_perm_b32:
// result[15:0] = flo[31:16], result[31:16] = fhi[31:16].
__device__ __forceinline__ unsigned pkbf(float flo, float fhi) {
  return __builtin_amdgcn_perm(__float_as_uint(fhi), __float_as_uint(flo),
                               0x07060302u);
}

__global__ __launch_bounds__(THREADS)
void fa_fwd_kernel(const float* __restrict__ Qg0, const float* __restrict__ Kg0,
                   const float* __restrict__ Vg0, float* __restrict__ Og) {
  __shared__ __bf16 sQ[BM * QLD];     //  49,664 B  (pre-scaled by 1/sqrt(D))
  __shared__ __bf16 sK[BN * KLD];     //  99,328 B
  __shared__ __bf16 sVt[D_ * VTLD];   // 110,592 B  (V transposed: [dim][key])
  __shared__ __bf16 sP[BM * PLD];     //   4,608 B  (probs, bf16)
  __shared__ float  sS[BM * SLD];     //   9,216 B  (scores, f32)
  __shared__ float  sM[BM];
  __shared__ float  sL[BM];
  __shared__ float  sA[BM];

  const int tid  = threadIdx.x;
  const int wave = tid >> 5;
  const int lane = tid & 31;
  const int hi   = lane >> 4;     // lane half (WMMA layout)
  const int nn   = lane & 15;     // N index within tile
  const int k0   = hi * 8;        // A-fragment K offset per ISA layout
  const int wm   = wave >> 2;     // query half  {0,1}
  const int wn   = wave & 3;      // QK: score-tile column {0..3}
  const int wd   = wave & 3;      // PV: 192-dim slice     {0..3}

  const int blk = blockIdx.x;
  const int bId = blk / (S_ / BM);
  const int m0  = (blk % (S_ / BM)) * BM;

  const float* Qg = Qg0 + ((size_t)bId * S_ + m0) * D_;
  const float* Kg = Kg0 + (size_t)bId * S_ * D_;
  const float* Vg = Vg0 + (size_t)bId * S_ * D_;

  const float inv_scale = 0.03608439182435161f;  // 1/sqrt(768)

  // ---- load Q tile (fp32 * inv_scale -> bf16, row-major) ----
  for (int i = tid; i < BM * (D_ / 4); i += THREADS) {
    int row = i / (D_ / 4);
    int c4  = i % (D_ / 4);
    const float4 q = *(const float4*)(Qg + (size_t)row * D_ + c4 * 4);
    uint2 p;
    p.x = pkbf(q.x * inv_scale, q.y * inv_scale);
    p.y = pkbf(q.z * inv_scale, q.w * inv_scale);
    *(uint2*)(void*)(sQ + row * QLD + c4 * 4) = p;
  }
  if (tid < BM) { sM[tid] = -3.0e38f; sL[tid] = 0.0f; sA[tid] = 0.0f; }

  // ---- output accumulators: 12 tiles of 16x16 f32 per wave ----
  v8f acc[NTD];
  const v8f vzero = {0.f,0.f,0.f,0.f,0.f,0.f,0.f,0.f};
#pragma unroll
  for (int t = 0; t < NTD; ++t) acc[t] = vzero;

  for (int kt = 0; kt < NKT; ++kt) {
    __syncthreads();  // previous iteration finished reading sK/sVt/sP

    // ---- stage K tile (row-major bf16, v_perm packing) ----
    for (int i = tid; i < BN * (D_ / 4); i += THREADS) {
      int key = i / (D_ / 4);
      int c4  = i % (D_ / 4);
      const float4 kv = *(const float4*)(Kg + (size_t)(kt * BN + key) * D_ + c4 * 4);
      uint2 p;
      p.x = pkbf(kv.x, kv.y);
      p.y = pkbf(kv.z, kv.w);
      *(uint2*)(void*)(sK + key * KLD + c4 * 4) = p;
    }
    // ---- stage V tile transposed: sVt[dim][key]; two keys per thread so each
    //      v_perm yields two adjacent columns -> single b32 LDS store ----
    for (int i = tid; i < (BN / 2) * (D_ / 4); i += THREADS) {
      int kp = i / (D_ / 4);            // key pair index 0..31
      int c4 = i % (D_ / 4);
      const float4 v0 = *(const float4*)(Vg + (size_t)(kt * BN + 2 * kp)     * D_ + c4 * 4);
      const float4 v1 = *(const float4*)(Vg + (size_t)(kt * BN + 2 * kp + 1) * D_ + c4 * 4);
      int dim = c4 * 4;
      *(unsigned*)(void*)(sVt + (dim + 0) * VTLD + 2 * kp) = pkbf(v0.x, v1.x);
      *(unsigned*)(void*)(sVt + (dim + 1) * VTLD + 2 * kp) = pkbf(v0.y, v1.y);
      *(unsigned*)(void*)(sVt + (dim + 2) * VTLD + 2 * kp) = pkbf(v0.z, v1.z);
      *(unsigned*)(void*)(sVt + (dim + 3) * VTLD + 2 * kp) = pkbf(v0.w, v1.w);
    }
    if (kt + 1 < NKT) {  // cover latency of next tile (global_prefetch_b8)
      __builtin_prefetch(Kg + (size_t)((kt + 1) * BN + (tid & 63)) * D_, 0, 0);
      __builtin_prefetch(Vg + (size_t)((kt + 1) * BN + (tid & 63)) * D_, 0, 0);
    }
    __syncthreads();

    // ---- QK^T: one 16x16 score tile per wave, 24 WMMAs over D ----
    {
      v8f c = vzero;
      const __bf16* qrow = sQ + (wm * 16 + nn) * QLD;  // A row = lane&15
      const __bf16* krow = sK + (wn * 16 + nn) * KLD;  // B col (key) = lane&15
#pragma unroll
      for (int dk = 0; dk < NDK; ++dk) {
        const int kb = dk * 32;
        v8bf alo = *(const v8bf*)(qrow + kb + k0);
        v8bf ahi = *(const v8bf*)(qrow + kb + 16 + k0);
        v16bf a  = CAT16(alo, ahi);
        v8bf blo = *(const v8bf*)(krow + kb + k0);
        v8bf bhi = *(const v8bf*)(krow + kb + 16 + k0);
        v16bf b  = CAT16(blo, bhi);
        c = __builtin_amdgcn_wmma_f32_16x16x32_bf16(false, a, false, b,
                                                    (short)0, c, false, false);
      }
#pragma unroll
      for (int r = 0; r < 8; ++r)
        sS[(wm * 16 + r + 8 * hi) * SLD + wn * 16 + nn] = c[r];
    }
    __syncthreads();

    // ---- online softmax: 8 threads per row, shuffle reductions ----
    {
      const int row = tid >> 3;
      const int sub = tid & 7;
      const float* srow = sS + row * SLD + sub * 8;
      float v[8];
      float mloc = -3.0e38f;
#pragma unroll
      for (int j = 0; j < 8; ++j) { v[j] = srow[j]; mloc = fmaxf(mloc, v[j]); }
#pragma unroll
      for (int o = 1; o < 8; o <<= 1) mloc = fmaxf(mloc, __shfl_xor(mloc, o, 8));
      const float mold = sM[row];
      const float mnew = fmaxf(mold, mloc);
      const float alpha = __expf(mold - mnew);
      float p[8];
      float sum = 0.f;
#pragma unroll
      for (int j = 0; j < 8; ++j) {
        p[j] = __expf(v[j] - mnew);
        sum += p[j];
      }
#pragma unroll
      for (int j = 0; j < 8; j += 2)
        *(unsigned*)(void*)(sP + row * PLD + sub * 8 + j) = pkbf(p[j], p[j + 1]);
#pragma unroll
      for (int o = 1; o < 8; o <<= 1) sum += __shfl_xor(sum, o, 8);
      if (sub == 0) {
        sM[row] = mnew;
        sL[row] = alpha * sL[row] + sum;
        sA[row] = alpha;
      }
    }
    __syncthreads();

    // ---- rescale accumulators + PV: 24 WMMAs per wave ----
    {
      float al[8];
#pragma unroll
      for (int r = 0; r < 8; ++r) al[r] = sA[wm * 16 + r + 8 * hi];
#pragma unroll
      for (int t = 0; t < NTD; ++t)
#pragma unroll
        for (int r = 0; r < 8; ++r) acc[t][r] *= al[r];

      const __bf16* prow = sP + (wm * 16 + nn) * PLD;  // A row = query
#pragma unroll
      for (int kk = 0; kk < 2; ++kk) {                 // two K=32 steps over 64 keys
        v8bf alo = *(const v8bf*)(prow + kk * 32 + k0);
        v8bf ahi = *(const v8bf*)(prow + kk * 32 + 16 + k0);
        v16bf a  = CAT16(alo, ahi);
#pragma unroll
        for (int t = 0; t < NTD; ++t) {
          const __bf16* vtrow = sVt + (wd * 192 + t * 16 + nn) * VTLD;  // B col = dim
          v8bf blo = *(const v8bf*)(vtrow + kk * 32 + k0);
          v8bf bhi = *(const v8bf*)(vtrow + kk * 32 + 16 + k0);
          v16bf b  = CAT16(blo, bhi);
          acc[t] = __builtin_amdgcn_wmma_f32_16x16x32_bf16(false, a, false, b,
                                                           (short)0, acc[t],
                                                           false, false);
        }
      }
    }
  }

  __syncthreads();  // sL stable

  // ---- finalize: divide by row sums, store fp32 ----
  {
    float invl[8];
#pragma unroll
    for (int r = 0; r < 8; ++r) invl[r] = 1.0f / sL[wm * 16 + r + 8 * hi];
    const size_t obase = ((size_t)bId * S_ + m0) * D_;
#pragma unroll
    for (int t = 0; t < NTD; ++t) {
      const int col = wd * 192 + t * 16 + nn;
#pragma unroll
      for (int r = 0; r < 8; ++r) {
        const int row = wm * 16 + r + 8 * hi;
        Og[obase + (size_t)row * D_ + col] = acc[t][r] * invl[r];
      }
    }
  }
}

extern "C" void kernel_launch(void* const* d_in, const int* in_sizes, int n_in,
                              void* d_out, int out_size, void* d_ws, size_t ws_size,
                              hipStream_t stream) {
  const float* x1 = (const float*)d_in[0];  // queries [B,S,D]
  const float* x2 = (const float*)d_in[1];  // keys    [B,S,D]
  const float* x3 = (const float*)d_in[2];  // values  [B,S,D]
  float* out = (float*)d_out;               // [B,S,D] fp32

  dim3 grid(B_ * (S_ / BM));
  dim3 block(THREADS);
  fa_fwd_kernel<<<grid, block, 0, stream>>>(x1, x2, x3, out);
}
